// Model_42743514530532
// MI455X (gfx1250) — matrix operand order repaired and checked by reading, not compiled, not run on hardware
//
#include <hip/hip_runtime.h>

typedef __attribute__((ext_vector_type(16))) _Float16 v16h;
typedef __attribute__((ext_vector_type(8)))  _Float16 v8h;
typedef __attribute__((ext_vector_type(8)))  float    v8f;
typedef __attribute__((ext_vector_type(4)))  float    f4;
typedef __attribute__((ext_vector_type(4)))  unsigned int u32x4;
typedef __attribute__((ext_vector_type(8)))  int i32x8;
typedef __attribute__((ext_vector_type(4)))  int i32x4;

#define NTOK        262144
#define HDIM        256
#define NEXP        4
#define TOK_PER_BLK 256
#define TOKW        32        // tokens per wave (2 row-tiles of 16)
#define HSTRIDE     264       // 256 + 8 halves pad -> conflict-free ds_load_b128
#define WROWS       128       // output cols per weight pass-tile
#define WTILE       (WROWS * HSTRIDE)   // halves per staged tile (padded)

#ifndef __has_builtin
#define __has_builtin(x) 0
#endif
#if __has_builtin(__builtin_amdgcn_tensor_load_to_lds) && __has_builtin(__builtin_amdgcn_s_wait_tensorcnt)
#define USE_TDM 1
#else
#define USE_TDM 0
#endif

// ---------------------------------------------------------------------------
// Prep: W1T[n][k] = (f16)W1[k][n]; WeT[e][f][h] = (f16)We[e][h][f]
// ---------------------------------------------------------------------------
__global__ void prep_weights(const float* __restrict__ W1,
                             const float* __restrict__ We,
                             _Float16* __restrict__ W1T,
                             _Float16* __restrict__ WeT) {
  int i = blockIdx.x * 256 + threadIdx.x;          // grid covers 262144
  if (i < 262144) {
    int e = i >> 16, rem = i & 65535;
    int f = rem >> 8, h = rem & 255;
    WeT[(size_t)e * 65536 + f * 256 + h] = (_Float16)We[(size_t)e * 65536 + h * 256 + f];
  }
  if (i < 65536) {
    int n = i >> 8, k = i & 255;
    W1T[n * 256 + k] = (_Float16)W1[k * 256 + n];
  }
}

// ---------------------------------------------------------------------------
// Stage one 64KB weight tile (128 rows x 256 K halves, contiguous in global)
// into LDS with a 16B pad per 512B row (-> stride HSTRIDE halves).
// TDM path: single tensor_load_to_lds issued by wave 0, tracked by TENSORcnt.
// ---------------------------------------------------------------------------
__device__ __forceinline__ void stage_tile(_Float16* lds_dst,
                                           const _Float16* src, int wave_id) {
#if USE_TDM
  if (wave_id == 0) {
    unsigned lds_off = (unsigned)(size_t)lds_dst;       // low 32 bits = LDS byte offset
    unsigned long long ga = (unsigned long long)(size_t)src;
    u32x4 g0;
    g0[0] = 1u;                                         // count=1, user descriptor
    g0[1] = lds_off;                                    // lds_addr (bytes)
    g0[2] = (unsigned)(ga & 0xffffffffull);             // global_addr[31:0]
    g0[3] = (unsigned)((ga >> 32) & 0x01ffffffull) | (2u << 30); // addr[56:32]|type=2
    i32x8 g1;
    // data_size=3 (8B units), pad_enable, pad_interval=6 (128 dw), pad_amount=3 (4 dw)
    g1[0] = (3 << 16) | (1 << 20) | (6 << 22) | (3 << 25);
    g1[1] = (int)0x20000000;    // tensor_dim0 = 0x2000 (8192 x 8B = 64KB), low16<<16
    g1[2] = (int)0x00010000;    // tensor_dim0 hi=0 | tensor_dim1 = 1
    g1[3] = (int)0x20000000;    // tensor_dim1 hi=0 | tile_dim0 = 0x2000
    g1[4] = 1;                  // tile_dim1 = 1, tile_dim2 = 0
    g1[5] = 0x2000;             // tensor_dim0_stride low32
    g1[6] = (int)0x20000000;    // stride0 hi=0 | tensor_dim1_stride low16 = 0x2000
    g1[7] = 0;
    i32x4 z4; z4[0] = z4[1] = z4[2] = z4[3] = 0;
    i32x8 z8; z8[0] = z8[1] = z8[2] = z8[3] = z8[4] = z8[5] = z8[6] = z8[7] = 0;
    __builtin_amdgcn_tensor_load_to_lds(g0, g1, z4, z4, z8, 0);
  }
#else
  // cooperative fallback: copy 4096 x 16B chunks, inserting the row pad
  const u32x4* s = (const u32x4*)src;
  for (int u = threadIdx.x; u < 4096; u += blockDim.x) {
    int row = u >> 5, cin = u & 31;                     // 32 chunks per 512B row
    *(u32x4*)((char*)lds_dst + row * (HSTRIDE * 2) + cin * 16) = s[u];
  }
#endif
}

template <int PENDING>
__device__ __forceinline__ void wait_tile() {
#if USE_TDM
  __builtin_amdgcn_s_wait_tensorcnt(PENDING);           // no-op for waves with cnt 0
#endif
  __syncthreads();
}

// ---------------------------------------------------------------------------
// Fused MoE, 8 waves/block, 32 tokens/wave, double-buffered TDM weight tiles
// ---------------------------------------------------------------------------
__global__ __launch_bounds__(256) void moe_kernel(
    const float* __restrict__ x, const float* __restrict__ b1,
    const float* __restrict__ Wg, const float* __restrict__ be,
    const _Float16* __restrict__ W1T, const _Float16* __restrict__ WeT,
    float* __restrict__ out)
{
  __shared__ _Float16 hsh[TOK_PER_BLK * HSTRIDE];   // h tile, f16 (132 KB)
  __shared__ _Float16 wbuf[2 * WTILE];              // double-buffered weights (132 KB)
  __shared__ float    gsh[TOK_PER_BLK];
  __shared__ int      ish[TOK_PER_BLK];

  const int lane  = threadIdx.x & 31;
  const int wave  = threadIdx.x >> 5;
  const int laneq = lane & 15;
  const bool hi   = lane >= 16;
  const int kbA   = hi ? 8 : 0;      // A frag: K {0..7,16..23} lo / {8..15,24..31} hi
  const int kbB   = hi ? 8 : 0;      // B frag: K {0..7,16..23} lo / {8..15,24..31} hi
  const int rsel  = hi ? 8 : 0;      // C frag row select
  const int tokBase = blockIdx.x * TOK_PER_BLK;
  const int wtok  = wave * TOKW;

  // Prefetch both stage-A weight pass-tiles (tiles T0,T1)
  stage_tile(wbuf,         W1T,         wave);
  stage_tile(wbuf + WTILE, W1T + 32768, wave);

  // ================= Stage A: h = x @ W1 + b1 =================
  const float* xr0 = x + (size_t)(tokBase + wtok + laneq) * HDIM;
  const float* xr1 = xr0 + 16 * HDIM;
  for (int p = 0; p < 2; ++p) {
    wait_tile<1>();                                 // tile p ready (next may be in flight)
    const _Float16* wb = wbuf + p * WTILE;
    v8f acc0[8] = {};
    v8f acc1[8] = {};
    for (int kc = 0; kc < 8; ++kc) {
      const int k0 = kc * 32;
      v16h a0, a1;
      {
        f4 t0 = *(const f4*)(xr0 + k0 + kbA);
        f4 t1 = *(const f4*)(xr0 + k0 + kbA + 4);
        f4 t2 = *(const f4*)(xr0 + k0 + kbA + 16);
        f4 t3 = *(const f4*)(xr0 + k0 + kbA + 20);
        f4 s0 = *(const f4*)(xr1 + k0 + kbA);
        f4 s1 = *(const f4*)(xr1 + k0 + kbA + 4);
        f4 s2 = *(const f4*)(xr1 + k0 + kbA + 16);
        f4 s3 = *(const f4*)(xr1 + k0 + kbA + 20);
        #pragma unroll
        for (int j = 0; j < 4; ++j) {
          a0[j] = (_Float16)t0[j]; a0[4+j]  = (_Float16)t1[j];
          a0[8+j] = (_Float16)t2[j]; a0[12+j] = (_Float16)t3[j];
          a1[j] = (_Float16)s0[j]; a1[4+j]  = (_Float16)s1[j];
          a1[8+j] = (_Float16)s2[j]; a1[12+j] = (_Float16)s3[j];
        }
      }
      #pragma unroll
      for (int nt = 0; nt < 8; ++nt) {
        v8h bl = *(const v8h*)(wb + (nt * 16 + laneq) * HSTRIDE + k0 + kbB);
        v8h bh = *(const v8h*)(wb + (nt * 16 + laneq) * HSTRIDE + k0 + kbB + 16);
        v16h b = __builtin_shufflevector(bl, bh, 0,1,2,3,4,5,6,7,8,9,10,11,12,13,14,15);
        acc0[nt] = __builtin_amdgcn_wmma_f32_16x16x32_f16(
            false, a0, false, b, (short)0, acc0[nt], false, false);
        acc1[nt] = __builtin_amdgcn_wmma_f32_16x16x32_f16(
            false, a1, false, b, (short)0, acc1[nt], false, false);
      }
    }
    // bias + stash h tile (f16) to LDS
    #pragma unroll
    for (int nt = 0; nt < 8; ++nt) {
      const int col = p * 128 + nt * 16 + laneq;
      const float bv = b1[col];
      #pragma unroll
      for (int i = 0; i < 8; ++i) {
        hsh[(wtok + i + rsel) * HSTRIDE + col]      = (_Float16)(acc0[nt][i] + bv);
        hsh[(wtok + 16 + i + rsel) * HSTRIDE + col] = (_Float16)(acc1[nt][i] + bv);
      }
    }
    __syncthreads();                                // buffer p free for reuse
    // issue stage-B tile j=p : (pass0, expert p) into freed buffer
    stage_tile(wbuf + p * WTILE, WeT + (size_t)p * 65536, wave);
  }

  // ================= Gating: one token per lane, wave-local =================
  {
    const int t = wtok + lane;
    float lg0 = 0.f, lg1 = 0.f, lg2 = 0.f, lg3 = 0.f;
    for (int k = 0; k < HDIM; ++k) {
      const float hv = (float)hsh[t * HSTRIDE + k];
      const f4 w = *(const f4*)(Wg + k * 4);
      lg0 += hv * w[0]; lg1 += hv * w[1]; lg2 += hv * w[2]; lg3 += hv * w[3];
    }
    float m = lg0; int idx = 0;
    if (lg1 > m) { m = lg1; idx = 1; }
    if (lg2 > m) { m = lg2; idx = 2; }
    if (lg3 > m) { m = lg3; idx = 3; }
    const float s = __expf(lg0-m) + __expf(lg1-m) + __expf(lg2-m) + __expf(lg3-m);
    gsh[t] = 1.0f / s;                 // prob of argmax expert
    ish[t] = idx;
  }

  // ===== Stage B: out = Sum_e (mask_e*g*h) @ We[e] + g*be[idx] =====
  const int rA0 = wtok + laneq;
  const int rA1 = wtok + 16 + laneq;
  for (int p = 0; p < 2; ++p) {
    v8f acc0[8] = {};
    v8f acc1[8] = {};
    for (int e = 0; e < NEXP; ++e) {
      const int j = p * 4 + e;                      // stage-B tile index 0..7
      if (j == 7) wait_tile<0>(); else wait_tile<1>();
      const _Float16* wb = wbuf + (j & 1) * WTILE;
      const _Float16 ge0 = (_Float16)((ish[rA0] == e) ? gsh[rA0] : 0.0f);
      const _Float16 ge1 = (_Float16)((ish[rA1] == e) ? gsh[rA1] : 0.0f);
      for (int kc = 0; kc < 8; ++kc) {
        const int k0 = kc * 32;
        v8h u0 = *(const v8h*)(hsh + rA0 * HSTRIDE + k0 + kbA);
        v8h u1 = *(const v8h*)(hsh + rA0 * HSTRIDE + k0 + kbA + 16);
        v8h u2 = *(const v8h*)(hsh + rA1 * HSTRIDE + k0 + kbA);
        v8h u3 = *(const v8h*)(hsh + rA1 * HSTRIDE + k0 + kbA + 16);
        v16h a0, a1;
        #pragma unroll
        for (int q = 0; q < 8; ++q) {
          a0[q] = u0[q] * ge0; a0[8+q] = u1[q] * ge0;
          a1[q] = u2[q] * ge1; a1[8+q] = u3[q] * ge1;
        }
        #pragma unroll
        for (int nt = 0; nt < 8; ++nt) {
          v8h bl = *(const v8h*)(wb + (nt * 16 + laneq) * HSTRIDE + k0 + kbB);
          v8h bh = *(const v8h*)(wb + (nt * 16 + laneq) * HSTRIDE + k0 + kbB + 16);
          v16h b = __builtin_shufflevector(bl, bh, 0,1,2,3,4,5,6,7,8,9,10,11,12,13,14,15);
          acc0[nt] = __builtin_amdgcn_wmma_f32_16x16x32_f16(
              false, a0, false, b, (short)0, acc0[nt], false, false);
          acc1[nt] = __builtin_amdgcn_wmma_f32_16x16x32_f16(
              false, a1, false, b, (short)0, acc1[nt], false, false);
        }
      }
      __syncthreads();                              // buffer (j&1) free
      if (j + 2 <= 7) {
        const int jn = j + 2;                       // tile jn = (pass jn/4, expert jn%4)
        stage_tile(wbuf + (jn & 1) * WTILE,
                   WeT + (size_t)(jn % 4) * 65536 + (size_t)(jn / 4) * 32768, wave);
      }
    }
    // gated expert bias + store f32 output
    float g0v[8], g1v[8]; int i0v[8], i1v[8];
    #pragma unroll
    for (int i = 0; i < 8; ++i) {
      g0v[i] = gsh[wtok + i + rsel];       i0v[i] = ish[wtok + i + rsel];
      g1v[i] = gsh[wtok + 16 + i + rsel];  i1v[i] = ish[wtok + 16 + i + rsel];
    }
    #pragma unroll
    for (int nt = 0; nt < 8; ++nt) {
      const int col = p * 128 + nt * 16 + laneq;
      #pragma unroll
      for (int i = 0; i < 8; ++i) {
        out[(size_t)(tokBase + wtok + i + rsel) * HDIM + col] =
            acc0[nt][i] + g0v[i] * be[i0v[i] * HDIM + col];
        out[(size_t)(tokBase + wtok + 16 + i + rsel) * HDIM + col] =
            acc1[nt][i] + g1v[i] * be[i1v[i] * HDIM + col];
      }
    }
  }
}

// ---------------------------------------------------------------------------
extern "C" void kernel_launch(void* const* d_in, const int* in_sizes, int n_in,
                              void* d_out, int out_size, void* d_ws, size_t ws_size,
                              hipStream_t stream) {
  const float* x  = (const float*)d_in[0];
  const float* W1 = (const float*)d_in[1];
  const float* b1 = (const float*)d_in[2];
  const float* Wg = (const float*)d_in[3];
  const float* We = (const float*)d_in[4];
  const float* be = (const float*)d_in[5];
  float* out = (float*)d_out;

  _Float16* W1T = (_Float16*)d_ws;            // 65536  f16 = 128 KB
  _Float16* WeT = W1T + 65536;                // 262144 f16 = 512 KB

  prep_weights<<<1024, 256, 0, stream>>>(W1, We, W1T, WeT);
  moe_kernel<<<NTOK / TOK_PER_BLK, 256, 0, stream>>>(x, b1, Wg, be, W1T, WeT, out);
}
